// GatedDeltaRuleModule_82514911691032
// MI455X (gfx1250) — matrix operand order repaired and checked by reading, not compiled, not run on hardware
//
#include <hip/hip_runtime.h>
#include <math.h>

// Gated sliding-window attention for MI455X (gfx1250, wave32, WMMA).
// One wave (32 threads) per (batch, head, 16-query tile).
// Phase 1: scores = Q.K^T over 3 aligned 16-key tiles via v_wmma_f32_16x16x32_f16.
// Gate+mask branchlessly in the 16x16 f32 C-layout, W stored f16 to LDS.
// Phase 2: out = W.V via WMMA with V pre-transposed (f16) in LDS.

typedef __attribute__((ext_vector_type(16))) _Float16 v16h;
typedef __attribute__((ext_vector_type(8)))  _Float16 v8h;
typedef __attribute__((ext_vector_type(8)))  float    v8f;
typedef __attribute__((ext_vector_type(4)))  float    v4f;

#define B_   2
#define S_   2048
#define H_   8
#define D_   64
#define HD_  (H_ * D_)
#define QT_  16
#define NT_  (S_ / QT_)

__launch_bounds__(32)
__global__ void swa_gated_wmma(const float* __restrict__ qg,
                               const float* __restrict__ kg,
                               const float* __restrict__ vg,
                               const float* __restrict__ ag,
                               const float* __restrict__ bg,
                               float* __restrict__ outg)
{
    // W[m][key_rel], key_rel in [0,64); keys 48..63 are zero padding.
    __shared__ __align__(16) _Float16 Wlds[16 * 64];
    // VT[channel][key_rel] = V[key][channel] (transposed), f16.
    __shared__ __align__(16) _Float16 VT[64 * 64];

    const int lane = threadIdx.x;   // single wave32
    const int lh   = lane & 15;
    const int hi   = lane >> 4;     // 0 or 1

    const int t  = blockIdx.x;
    const int qt = t % NT_;
    const int h  = (t / NT_) % H_;
    const int bb = t / (NT_ * H_);
    const int s0 = qt * QT_;

    // ---- zero the pad regions (key_rel 48..63) ----
    {
        v8h z = {};
        const int r0 = lane >> 1, p0 = lane & 1;          // 32 chunks cover 16 rows
        *reinterpret_cast<v8h*>(&Wlds[r0 * 64 + 48 + p0 * 8]) = z;
        for (int i = lane; i < 128; i += 32) {            // 64 rows x 2 chunks
            const int ch = i >> 1, p = i & 1;
            *reinterpret_cast<v8h*>(&VT[ch * 64 + 48 + p * 8]) = z;
        }
    }

    // ---- stage V window transposed into LDS as f16 ----
    // rows 0..47 <-> keys s0-32 .. s0+15 (negative keys contribute 0 anyway)
    for (int i = lane; i < 48 * 16; i += 32) {
        const int row = i >> 4;
        const int ck  = (i & 15) * 4;
        const int key = s0 - 32 + row;
        v4f val = {};
        if (key >= 0)
            val = *reinterpret_cast<const v4f*>(
                vg + (size_t)(bb * S_ + key) * HD_ + h * D_ + ck);
#pragma unroll
        for (int j = 0; j < 4; ++j)
            VT[(ck + j) * 64 + row] = (_Float16)val[j];
    }

    // ---- build Q A-operands (16x32 f16 chunks, contiguous k = 16*hi + s order) ----
    v16h qop[2];
    {
        const float* qrow = qg + (size_t)(bb * S_ + s0 + lh) * HD_ + h * D_;
#pragma unroll
        for (int c = 0; c < 2; ++c) {
            const int kb = 32 * c + hi * 16;
            v4f f0 = *reinterpret_cast<const v4f*>(qrow + kb);
            v4f f1 = *reinterpret_cast<const v4f*>(qrow + kb + 4);
            v4f f2 = *reinterpret_cast<const v4f*>(qrow + kb + 8);
            v4f f3 = *reinterpret_cast<const v4f*>(qrow + kb + 12);
#pragma unroll
            for (int e = 0; e < 4; ++e) {
                qop[c][e]      = (_Float16)f0[e];
                qop[c][4 + e]  = (_Float16)f1[e];
                qop[c][8 + e]  = (_Float16)f2[e];
                qop[c][12 + e] = (_Float16)f3[e];
            }
        }
    }

    // ---- per-query gate scalars b[s0+m] ----
    float bvals[8];
#pragma unroll
    for (int r = 0; r < 8; ++r)
        bvals[r] = bg[(size_t)(bb * S_ + s0 + r + 8 * hi) * H_ + h];

    const float scale = 0.125f;   // 1/sqrt(64)

    // ---- phase 1: scores per 16-key tile, gate, mask, store W (f16) ----
#pragma unroll
    for (int kt = 0; kt < 3; ++kt) {
        const int key_base = s0 - 32 + 16 * kt;   // uniform across wave
        v8f acc = {};
        if (key_base >= 0) {                      // scalar branch: EXEC stays full
            const float* krow =
                kg + (size_t)(bb * S_ + key_base + lh) * HD_ + h * D_;
#pragma unroll
            for (int c = 0; c < 2; ++c) {
                const int cb = 32 * c + hi * 16;  // same k order as Q: 16 contiguous ch
                v4f g0 = *reinterpret_cast<const v4f*>(krow + cb);
                v4f g1 = *reinterpret_cast<const v4f*>(krow + cb + 4);
                v4f g2 = *reinterpret_cast<const v4f*>(krow + cb + 8);
                v4f g3 = *reinterpret_cast<const v4f*>(krow + cb + 12);
                v16h kop;
#pragma unroll
                for (int e = 0; e < 4; ++e) {
                    kop[e]      = (_Float16)g0[e];
                    kop[4 + e]  = (_Float16)g1[e];
                    kop[8 + e]  = (_Float16)g2[e];
                    kop[12 + e] = (_Float16)g3[e];
                }
                acc = __builtin_amdgcn_wmma_f32_16x16x32_f16(
                    false, qop[c], false, kop, (short)0, acc, false, false);
            }
        }

        // branchless gate + window mask + store (reference clips the index too)
        const int key  = key_base + lh;
        const int keyc = key < 0 ? 0 : key;                 // v_max, no branch
        const float aval = ag[(size_t)(bb * S_ + keyc) * H_ + h];
        const int kr = 16 * kt + lh;                        // key_rel (C-layout col)

#pragma unroll
        for (int r = 0; r < 8; ++r) {
            const int m = r + 8 * hi;                       // query row (C layout)
            // in-window <=> 1 <= kr - m <= 32, plus key >= 0
            const bool valid =
                ((unsigned)(kr - m - 1) < 32u) && (key >= 0);
            // fast sigmoid: v_exp + v_rcp (plenty of precision over f16 operands)
            const float g =
                __builtin_amdgcn_rcpf(1.0f + __expf(-aval * bvals[r]));
            const float wv = valid ? acc[r] * scale * g : 0.0f;   // v_cndmask
            Wlds[m * 64 + kr] = (_Float16)wv;
        }
    }

    __syncthreads();   // single-wave WG: cheap; orders LDS W/VT for phase 2

    // ---- phase 2: out(16x64) += W(16x64) x V(64x64, zero-padded keys) ----
    v8f oacc[4];
#pragma unroll
    for (int dc = 0; dc < 4; ++dc) oacc[dc] = (v8f){};

#pragma unroll
    for (int c = 0; c < 2; ++c) {
        const int kb = 32 * c + hi * 16;      // A layout key offsets
        v8h alo = *reinterpret_cast<const v8h*>(&Wlds[lh * 64 + kb]);
        v8h ahi = *reinterpret_cast<const v8h*>(&Wlds[lh * 64 + kb + 8]);
        v16h aop = __builtin_shufflevector(alo, ahi,
            0, 1, 2, 3, 4, 5, 6, 7, 8, 9, 10, 11, 12, 13, 14, 15);

        const int kk = 32 * c + hi * 16;      // B layout key offsets
#pragma unroll
        for (int dc = 0; dc < 4; ++dc) {
            const int ch = dc * 16 + lh;
            v8h b0 = *reinterpret_cast<const v8h*>(&VT[ch * 64 + kk]);
            v8h b1 = *reinterpret_cast<const v8h*>(&VT[ch * 64 + kk + 8]);
            v16h bop = __builtin_shufflevector(b0, b1,
                0, 1, 2, 3, 4, 5, 6, 7, 8, 9, 10, 11, 12, 13, 14, 15);
            oacc[dc] = __builtin_amdgcn_wmma_f32_16x16x32_f16(
                false, aop, false, bop, (short)0, oacc[dc], false, false);
        }
    }

    // ---- store out (fp32, C layout -> memory) ----
#pragma unroll
    for (int dc = 0; dc < 4; ++dc) {
#pragma unroll
        for (int r = 0; r < 8; ++r) {
            const int m = r + 8 * hi;
            outg[(size_t)(bb * S_ + s0 + m) * HD_ + h * D_ + dc * 16 + lh] =
                oacc[dc][r];
        }
    }
}

extern "C" void kernel_launch(void* const* d_in, const int* in_sizes, int n_in,
                              void* d_out, int out_size, void* d_ws, size_t ws_size,
                              hipStream_t stream)
{
    const float* q = (const float*)d_in[0];
    const float* k = (const float*)d_in[1];
    const float* v = (const float*)d_in[2];
    const float* a = (const float*)d_in[3];
    const float* b = (const float*)d_in[4];
    float* out = (float*)d_out;

    dim3 grid(B_ * H_ * NT_);   // 2048 tiles
    dim3 block(32);             // one wave32 per tile
    swa_gated_wmma<<<grid, block, 0, stream>>>(q, k, v, a, b, out);
}
